// InContext_Predict_60894046322849
// MI455X (gfx1250) — matrix operand order repaired and checked by reading, not compiled, not run on hardware
//
#include <hip/hip_runtime.h>
#include <hip/hip_bf16.h>

// ---------------------------------------------------------------------------
// InContext KNN predict for MI455X (gfx1250, wave32, WMMA f16 16x16x32).
//   sim = normalize(x) @ normalize(ref_x)^T   (2048 x 100000, K=256)
//   top-100 per row -> weights -> weighted sum of ref_y rows.
// Streaming top-k: sim never materialized. Each wave owns 32 x-rows (two
// 16-row WMMA blocks). B tiles are staged through LDS with double-buffered
// GLOBAL_LOAD_ASYNC_TO_LDS_B128 (ASYNCcnt), hiding L2 latency behind a full
// tile of WMMA work with zero extra VGPR cost.
// ---------------------------------------------------------------------------

typedef _Float16 half8 __attribute__((ext_vector_type(8)));
typedef _Float16 v16h  __attribute__((ext_vector_type(16)));
typedef float    v8f   __attribute__((ext_vector_type(8)));

#define N_ROWS  2048
#define H_DIM   256
#define R_ROWS  100000
#define C_DIM   64
#define KKEEP   100
#define SPLIT   16
#define KSTEPS  (H_DIM / 32)            // 8 WMMA k-steps of 32
#define TILES_TOTAL (R_ROWS / 16)       // 6250 column tiles
#define TPC ((TILES_TOTAL + SPLIT - 1) / SPLIT)  // 391 tiles per chunk
#define MROWS   32                      // x-rows per wave (two 16-row blocks)
#define BROW_H  264                     // LDS B-row stride in halves (528B: bank-shift 4)

// Async copy: LDS[lds_off] = MEM[gaddr], 16B per lane, tracked by ASYNCcnt.
__device__ __forceinline__ void async_b128_to_lds(unsigned lds_off, const void* gaddr) {
  asm volatile("global_load_async_to_lds_b128 %0, %1, off"
               :: "v"(lds_off), "v"((unsigned long long)(uintptr_t)gaddr)
               : "memory");
}
#define WAIT_ASYNCCNT(n) asm volatile("s_wait_asynccnt " #n ::: "memory")

// ---------------------------------------------------------------------------
// Kernel 1: L2-normalize rows of a [rows x 256] f32 matrix into f16.
// One block (256 threads) per row.
// ---------------------------------------------------------------------------
__global__ __launch_bounds__(256)
void knn_normalize_rows(const float* __restrict__ in, _Float16* __restrict__ out) {
  const int row = blockIdx.x;
  const int tid = threadIdx.x;
  float v = in[(size_t)row * H_DIM + tid];
  float ss = v * v;
  for (int off = 16; off > 0; off >>= 1) ss += __shfl_xor(ss, off, 32);
  __shared__ float red[8];
  if ((tid & 31) == 0) red[tid >> 5] = ss;
  __syncthreads();
  float tot = 0.f;
  #pragma unroll
  for (int i = 0; i < 8; ++i) tot += red[i];
  float inv = 1.0f / sqrtf(tot);
  out[(size_t)row * H_DIM + tid] = (_Float16)(v * inv);
}

// ---------------------------------------------------------------------------
// Kernel 2: WMMA GEMM + streaming per-row top-100 over a chunk of ref rows.
// Grid: (N_ROWS/32, SPLIT). Block: 1 wave (32 threads). Lane l owns row l.
// Writes sorted-ascending candidate lists [row][chunk][KKEEP] to scratch.
// ---------------------------------------------------------------------------
__global__ __launch_bounds__(32)
void knn_gemm_topk(const _Float16* __restrict__ xn, const _Float16* __restrict__ rn,
                   float* __restrict__ cand_val, int* __restrict__ cand_idx) {
  const int rowblk = blockIdx.x;        // 32-row block of x
  const int chunk  = blockIdx.y;        // chunk of ref rows
  const int lane   = threadIdx.x;       // 0..31
  const int mrow   = lane & 15;
  const int hi     = lane >> 4;         // 0 or 1

  __shared__ float    s_val[MROWS][KKEEP]; // sorted ascending top-100 per row
  __shared__ int      s_idx[MROWS][KKEEP];
  __shared__ float    s_th[MROWS];         // current 100th-largest per row
  __shared__ float    s_tile[MROWS][21];   // 32x16 sim tile (stride 21: conflict-free)
  __shared__ _Float16 s_b[2 * 16 * BROW_H]; // double-buffered B tile (16 rows x 528B)

  for (int i = lane; i < MROWS * KKEEP; i += 32) {
    (&s_val[0][0])[i] = -3.0e38f;
    (&s_idx[0][0])[i] = 0;
  }
  s_th[lane] = -3.0e38f;
  __syncthreads();

  // --- Persistent A fragments: 32 x-rows, all K=256 (128 VGPRs) -----------
  // A layout (16-bit 16x32): lane m=lane&15; halves 0..7 -> K = kb+8*hi+j,
  // halves 8..15 -> K = kb+16+8*hi+j.
  const _Float16* arow0 = xn + (size_t)(rowblk * MROWS + mrow) * H_DIM;
  const _Float16* arow1 = arow0 + (size_t)16 * H_DIM;
  v16h a0[KSTEPS], a1[KSTEPS];
  #pragma unroll
  for (int ks = 0; ks < KSTEPS; ++ks) {
    const int kb = ks * 32;
    half8 l0 = *(const half8*)(arow0 + kb + hi * 8);
    half8 h0 = *(const half8*)(arow0 + kb + 16 + hi * 8);
    a0[ks] = __builtin_shufflevector(l0, h0, 0,1,2,3,4,5,6,7,8,9,10,11,12,13,14,15);
    half8 l1 = *(const half8*)(arow1 + kb + hi * 8);
    half8 h1 = *(const half8*)(arow1 + kb + 16 + hi * 8);
    a1[ks] = __builtin_shufflevector(l1, h1, 0,1,2,3,4,5,6,7,8,9,10,11,12,13,14,15);
  }

  // Issue one B tile (8KB) as 16 async row-copies: row r -> s_b[buf] row r,
  // each lane moving 16B. LDS row stride 528B keeps later ds_loads conflict-free.
  auto issue_tile = [&](int t, int buf) {
    const _Float16* g0 = rn + ((size_t)t * 16) * H_DIM + lane * 8;
    const unsigned  l0 = (unsigned)(uintptr_t)(s_b + buf * (16 * BROW_H) + lane * 8);
    #pragma unroll
    for (int r = 0; r < 16; ++r)
      async_b128_to_lds(l0 + r * (BROW_H * 2), g0 + (size_t)r * H_DIM);
  };

  // threshold register cache: thc[v] guards acc0 row v+8*hi, thc[8+v] guards
  // acc1 row 16+v+8*hi.
  float thc[16];
  #pragma unroll
  for (int v = 0; v < 8; ++v) {
    thc[v]     = s_th[v + 8 * hi];
    thc[8 + v] = s_th[16 + v + 8 * hi];
  }

  const int t0 = chunk * TPC;
  const int t1 = (t0 + TPC < TILES_TOTAL) ? (t0 + TPC) : TILES_TOTAL;

  issue_tile(t0, 0);                     // prime the pipeline

  for (int t = t0; t < t1; ++t) {
    const int buf = (t - t0) & 1;
    if (t + 1 < t1) {
      issue_tile(t + 1, buf ^ 1);        // 16 newer copies in flight
      WAIT_ASYNCCNT(16);                 // older 16 (this tile) retired
    } else {
      WAIT_ASYNCCNT(0);
    }

    // B fragments from LDS. B layout (16-bit 32x16): lane n=lane&15; halves 0..7
    // -> K = kb+8*hi+j, halves 8..15 -> K = kb+16+8*hi+j (matches A).
    const _Float16* bbase = s_b + buf * (16 * BROW_H) + mrow * BROW_H;
    v8f acc0 = {}, acc1 = {};
    #pragma unroll
    for (int ks = 0; ks < KSTEPS; ++ks) {
      half8 bl = *(const half8*)(bbase + ks * 32 + hi * 8);
      half8 bh = *(const half8*)(bbase + ks * 32 + 16 + hi * 8);
      v16h bf = __builtin_shufflevector(bl, bh, 0,1,2,3,4,5,6,7,8,9,10,11,12,13,14,15);
      acc0 = __builtin_amdgcn_wmma_f32_16x16x32_f16(
          false, a0[ks], false, bf, (short)0, acc0, false, false);
      acc1 = __builtin_amdgcn_wmma_f32_16x16x32_f16(
          false, a1[ks], false, bf, (short)0, acc1, false, false);
    }

    // fast reject: does any lane beat its row thresholds?
    bool pass = false;
    #pragma unroll
    for (int v = 0; v < 8; ++v) {
      pass |= (acc0[v] > thc[v]);
      pass |= (acc1[v] > thc[8 + v]);
    }
    if (!__any((int)pass)) continue;     // wave-uniform (1 wave per WG)

    const int nbase = t * 16;
    // spill tile to LDS: acc0 -> rows 0..15, acc1 -> rows 16..31
    #pragma unroll
    for (int v = 0; v < 8; ++v) {
      s_tile[v + 8 * hi][mrow]      = acc0[v];
      s_tile[16 + v + 8 * hi][mrow] = acc1[v];
    }
    __syncthreads();

    {                                     // every lane owns one row: sorted insert
      float* rv = &s_val[lane][0];
      int*   ri = &s_idx[lane][0];
      float th = s_th[lane];
      for (int n = 0; n < 16; ++n) {
        float val = s_tile[lane][n];
        if (val > th) {
          int j = 0;
          while (j < KKEEP - 1 && rv[j + 1] < val) {
            rv[j] = rv[j + 1];
            ri[j] = ri[j + 1];
            ++j;
          }
          rv[j] = val;
          ri[j] = nbase + n;
          th = rv[0];
        }
      }
      s_th[lane] = th;
    }
    __syncthreads();
    #pragma unroll
    for (int v = 0; v < 8; ++v) {
      thc[v]     = s_th[v + 8 * hi];
      thc[8 + v] = s_th[16 + v + 8 * hi];
    }
  }

  __syncthreads();
  // write candidate lists: [row][chunk][KKEEP]
  for (int i = lane; i < MROWS * KKEEP; i += 32) {
    const int r = i / KKEEP, j = i % KKEEP;
    const size_t o = ((size_t)(rowblk * MROWS + r) * SPLIT + chunk) * KKEEP + j;
    cand_val[o] = s_val[r][j];
    cand_idx[o] = s_idx[r][j];
  }
}

// ---------------------------------------------------------------------------
// Kernel 3: per x-row: 16-way merge of sorted chunk lists -> global top-100,
// then weighted gather-sum of ref_y. One block (128 threads) per row.
// ---------------------------------------------------------------------------
__global__ __launch_bounds__(128)
void knn_merge_predict(const float* __restrict__ cand_val, const int* __restrict__ cand_idx,
                       const float* __restrict__ ref_y, float* __restrict__ out) {
  const int row = blockIdx.x;
  const int tid = threadIdx.x;

  __shared__ float mv[SPLIT * KKEEP];
  __shared__ int   mi[SPLIT * KKEEP];
  __shared__ float sv[KKEEP];
  __shared__ int   si[KKEEP];
  __shared__ float ssum;

  const float* cv = cand_val + (size_t)row * SPLIT * KKEEP;
  const int*   ci = cand_idx + (size_t)row * SPLIT * KKEEP;
  for (int i = tid; i < SPLIT * KKEEP; i += 128) { mv[i] = cv[i]; mi[i] = ci[i]; }
  __syncthreads();

  if (tid == 0) {                         // serial 16-way merge (lists ascending)
    int head[SPLIT];
    #pragma unroll
    for (int s = 0; s < SPLIT; ++s) head[s] = KKEEP - 1;
    float sum = 0.f;
    for (int k = 0; k < KKEEP; ++k) {
      float bv = -3.0e38f; int bs = 0;
      #pragma unroll
      for (int s = 0; s < SPLIT; ++s) {
        const int h = head[s];
        if (h >= 0) {
          const float v = mv[s * KKEEP + h];
          if (v > bv) { bv = v; bs = s; }
        }
      }
      sv[k] = bv;
      si[k] = mi[bs * KKEEP + head[bs]];
      head[bs]--;
      sum += bv;
    }
    ssum = sum;
  }
  __syncthreads();

  if (tid < C_DIM) {                      // coalesced weighted gather of ref_y
    float acc = 0.f;
    for (int k = 0; k < KKEEP; ++k)
      acc += sv[k] * ref_y[(size_t)si[k] * C_DIM + tid];
    out[(size_t)row * C_DIM + tid] = acc / ssum;
  }
}

// ---------------------------------------------------------------------------
extern "C" void kernel_launch(void* const* d_in, const int* in_sizes, int n_in,
                              void* d_out, int out_size, void* d_ws, size_t ws_size,
                              hipStream_t stream) {
  const float* x     = (const float*)d_in[0];   // [2048, 256]
  const float* ref_x = (const float*)d_in[1];   // [100000, 256]
  const float* ref_y = (const float*)d_in[2];   // [100000, 64]
  // d_in[3] = num_neighbors (==100, compile-time KKEEP)
  float* out = (float*)d_out;                   // [2048, 64]

  // scratch layout (all offsets 256B-aligned)
  char* ws = (char*)d_ws;
  _Float16* xn = (_Float16*)ws;                                 //  1.0 MB
  _Float16* rn = xn + (size_t)N_ROWS * H_DIM;                   // 51.2 MB
  float* cand_val = (float*)(rn + (size_t)R_ROWS * H_DIM);      // 13.1 MB
  int*   cand_idx = (int*)(cand_val + (size_t)N_ROWS * SPLIT * KKEEP); // 13.1 MB

  knn_normalize_rows<<<N_ROWS, 256, 0, stream>>>(x, xn);
  knn_normalize_rows<<<R_ROWS, 256, 0, stream>>>(ref_x, rn);
  knn_gemm_topk<<<dim3(N_ROWS / MROWS, SPLIT), 32, 0, stream>>>(xn, rn, cand_val, cand_idx);
  knn_merge_predict<<<N_ROWS, 128, 0, stream>>>(cand_val, cand_idx, ref_y, out);
}
